// TrainingFreeSelfAttention_74852690035310
// MI455X (gfx1250) — matrix operand order repaired and checked by reading, not compiled, not run on hardware
//
#include <hip/hip_runtime.h>

typedef __attribute__((ext_vector_type(16))) _Float16 v16h;
typedef __attribute__((ext_vector_type(8)))  _Float16 v8h;
typedef __attribute__((ext_vector_type(8)))  float    v8f;
typedef __attribute__((ext_vector_type(4)))  unsigned int v4u;
typedef __attribute__((ext_vector_type(8)))  int v8i;
typedef __attribute__((ext_vector_type(4)))  int v4i;

#define N_TOK  8192
#define CDIM   128
#define QWAVES 8
#define KC     64          // keys staged per chunk
#define KPAD   8           // LDS row padding in halves (16B) -> conflict-free ds_load_b128

__device__ __forceinline__ unsigned lds_off(const void* p) {
    // low 32 bits of a flat LDS pointer are the byte offset within the group's LDS
    return (unsigned)(unsigned long long)p;
}

// TDM: DMA a 2D tile (tile_d1 rows of tile_d0 elements, f16) global -> LDS,
// inserting KPAD*2 bytes of padding after every LDS row (w0pad configures it).
__device__ __forceinline__ void tdm_load_2d(unsigned lds_addr, const void* gptr,
                                            unsigned tile_d0, unsigned tile_d1,
                                            unsigned tensor_d0, unsigned tensor_d1,
                                            unsigned long long stride_d0,
                                            unsigned w0pad) {
    unsigned long long ga = (unsigned long long)gptr;
    v4u g0;
    g0.x = 1u;                                                // count=1, user descriptor
    g0.y = lds_addr;                                          // LDS byte address
    g0.z = (unsigned)(ga & 0xFFFFFFFFu);                      // global_addr[31:0]
    g0.w = (unsigned)((ga >> 32) & 0x1FFFFFFu) | (2u << 30);  // global_addr[56:32] | type=2
    v8i g1;
    g1[0] = (int)w0pad;                                       // data_size=2B | pad cfg
    g1[1] = (int)((tensor_d0 & 0xFFFFu) << 16);               // abar=0 | tensor_dim0 lo
    g1[2] = (int)(((tensor_d0 >> 16) & 0xFFFFu) | ((tensor_d1 & 0xFFFFu) << 16));
    g1[3] = (int)(((tensor_d1 >> 16) & 0xFFFFu) | ((tile_d0 & 0xFFFFu) << 16));
    g1[4] = (int)(tile_d1 & 0xFFFFu);                         // tile_dim1 | tile_dim2=0
    g1[5] = (int)(unsigned)(stride_d0 & 0xFFFFFFFFu);         // tensor_dim0_stride lo
    g1[6] = (int)(unsigned)((stride_d0 >> 32) & 0xFFFFu);     // stride hi | dim1_stride=0
    g1[7] = 0;
    v4i z4 = {0, 0, 0, 0};
    v8i z8 = {0, 0, 0, 0, 0, 0, 0, 0};
    __builtin_amdgcn_tensor_load_to_lds(g0, g1, z4, z4, z8, 0);
}

// word0 of D# group1: data_size=1 (2 bytes), pad_enable, pad_interval, pad_amount(=4 dwords)
#define W0_K 0x07510000u   // pad every 64 dwords (256B row) + 16B  -> 272B LDS row stride
#define W0_V 0x07110000u   // pad every 32 dwords (128B row) + 16B  -> 144B LDS row stride

// Pack NCHW f32 -> fN [N][C] f16 (token-major) and fT [C][N] f16 (channel-major).
__global__ void tfsa_pack_kernel(const float* __restrict__ in,
                                 _Float16* __restrict__ fN,
                                 _Float16* __restrict__ fT) {
    int idx = blockIdx.x * blockDim.x + threadIdx.x;   // [0, 2*128*4096)
    int hw  = idx & 4095;
    int c   = (idx >> 12) & 127;
    int b   = idx >> 19;
    _Float16 h = (_Float16)in[idx];
    int n = (b << 12) | hw;
    fN[n * CDIM + c]  = h;
    fT[(c << 13) + n] = h;
}

__global__ void __launch_bounds__(32 * QWAVES)
tfsa_attn_kernel(const _Float16* __restrict__ fN,
                 const _Float16* __restrict__ fT,
                 float* __restrict__ out) {
    // double-buffered K (token-major) and V^T (channel-major) staging tiles
    __shared__ __attribute__((aligned(16))) _Float16 Kbuf[2][KC][CDIM + KPAD];
    __shared__ __attribute__((aligned(16))) _Float16 Vbuf[2][CDIM][KC + KPAD];

    const int lane = threadIdx.x & 31;
    const int wave = threadIdx.x >> 5;
    const int row  = lane & 15;
    const int hi   = lane >> 4;
    const int qbase = (blockIdx.x * QWAVES + wave) << 4;   // 16 queries per wave

    const _Float16 SL = (_Float16)(0.08838834764831845f * 1.4426950408889634f);

    // Q^T fragments (B-operand layout), scale folded in
    v16h qf[4];
    const _Float16* qrow = fN + (size_t)(qbase + row) * CDIM;
#pragma unroll
    for (int c = 0; c < 4; ++c) {
        qf[c] = *(const v16h*)(qrow + c * 32 + hi * 16);
        qf[c] *= SL;
    }

    v8f o[8] = {};
    float m_run = -1.0e30f;
    float l_run = 0.0f;

    const int NCHUNK = N_TOK / KC;

    // prefetch chunk 0 into buffer 0
    if (wave == 0) {
        tdm_load_2d(lds_off(&Kbuf[0][0][0]), fN, CDIM, KC, CDIM, KC,
                    (unsigned long long)CDIM, W0_K);
        tdm_load_2d(lds_off(&Vbuf[0][0][0]), fT, KC, CDIM, N_TOK, CDIM,
                    (unsigned long long)N_TOK, W0_V);
        __builtin_amdgcn_s_wait_tensorcnt(0);
    }
    __syncthreads();

    for (int ch = 0; ch < NCHUNK; ++ch) {
        const int cur = ch & 1;

        // kick off DMA of the next chunk into the other buffer (overlaps compute)
        if (wave == 0 && (ch + 1) < NCHUNK) {
            const int k0n = (ch + 1) * KC;
            tdm_load_2d(lds_off(&Kbuf[cur ^ 1][0][0]), fN + (size_t)k0n * CDIM,
                        CDIM, KC, CDIM, KC, (unsigned long long)CDIM, W0_K);
            tdm_load_2d(lds_off(&Vbuf[cur ^ 1][0][0]), fT + k0n,
                        KC, CDIM, N_TOK, CDIM, (unsigned long long)N_TOK, W0_V);
        }

#pragma unroll
        for (int kk = 0; kk < KC; kk += 32) {
            // ---- S^T = K * Q^T for two 16-key tiles from LDS ----
            const _Float16* kr0 = &Kbuf[cur][kk + row][0];
            const _Float16* kr1 = &Kbuf[cur][kk + 16 + row][0];
            v8f st0 = {}, st1 = {};
#pragma unroll
            for (int c = 0; c < 4; ++c) {
                union { v16h v; v8h h[2]; } a;
                a.h[0] = *(const v8h*)(kr0 + c * 32 + hi * 16);
                a.h[1] = *(const v8h*)(kr0 + c * 32 + hi * 16 + 8);
                st0 = __builtin_amdgcn_wmma_f32_16x16x32_f16(false, a.v, false, qf[c],
                                                             (short)0, st0, false, false);
            }
#pragma unroll
            for (int c = 0; c < 4; ++c) {
                union { v16h v; v8h h[2]; } a;
                a.h[0] = *(const v8h*)(kr1 + c * 32 + hi * 16);
                a.h[1] = *(const v8h*)(kr1 + c * 32 + hi * 16 + 8);
                st1 = __builtin_amdgcn_wmma_f32_16x16x32_f16(false, a.v, false, qf[c],
                                                             (short)0, st1, false, false);
            }

            // ---- online softmax (base-2 domain; lane covers 16 of 32 keys) ----
            float mt = st0[0];
#pragma unroll
            for (int i = 1; i < 8; ++i) mt = fmaxf(mt, st0[i]);
#pragma unroll
            for (int i = 0; i < 8; ++i) mt = fmaxf(mt, st1[i]);
            mt = fmaxf(mt, __shfl_xor(mt, 16, 32));
            float m_new = fmaxf(m_run, mt);
            float corr  = __builtin_amdgcn_exp2f(m_run - m_new);

            union HU { v8h h; int i[4]; } h0, h1, g0, g1;
            float s = 0.0f;
#pragma unroll
            for (int i = 0; i < 8; ++i) {
                float p = __builtin_amdgcn_exp2f(st0[i] - m_new);
                s += p; h0.h[i] = (_Float16)p;
            }
#pragma unroll
            for (int i = 0; i < 8; ++i) {
                float p = __builtin_amdgcn_exp2f(st1[i] - m_new);
                s += p; h1.h[i] = (_Float16)p;
            }
            s += __shfl_xor(s, 16, 32);
            l_run = l_run * corr + s;
            m_run = m_new;

            // ---- build P^T B-fragment via half-wave swap ----
#pragma unroll
            for (int j = 0; j < 4; ++j) {
                g0.i[j] = __shfl_xor(h0.i[j], 16, 32);
                g1.i[j] = __shfl_xor(h1.i[j], 16, 32);
            }
            union { v16h v; v8h h[2]; } bp;
            bp.h[0] = hi ? g1.h : h0.h;
            bp.h[1] = hi ? h1.h : g0.h;

            // ---- O^T += V^T * P^T (V^T rows from LDS) ----
#pragma unroll
            for (int t = 0; t < 8; ++t) o[t] *= corr;
#pragma unroll
            for (int t = 0; t < 8; ++t) {
                const _Float16* vr = &Vbuf[cur][t * 16 + row][kk + hi * 16];
                union { v16h v; v8h h[2]; } a;
                a.h[0] = *(const v8h*)(vr);
                a.h[1] = *(const v8h*)(vr + 8);
                o[t] = __builtin_amdgcn_wmma_f32_16x16x32_f16(false, a.v, false, bp.v,
                                                              (short)0, o[t], false, false);
            }
        }

        __syncthreads();                                   // all waves done with buf[cur]
        if (wave == 0) __builtin_amdgcn_s_wait_tensorcnt(0);
        __syncthreads();                                   // buf[cur^1] ready for next iter
    }

    // ---- epilogue: normalize, scatter back to NCHW f32 ----
    float inv_l = 1.0f / l_run;
    int n  = qbase + row;
    int b  = n >> 12;
    int hw = n & 4095;
#pragma unroll
    for (int t = 0; t < 8; ++t) {
#pragma unroll
        for (int r = 0; r < 8; ++r) {
            int cg = t * 16 + r + hi * 8;   // D layout: VGPR r -> channel row m = r + 8*hi
            out[(size_t)((b * CDIM + cg) << 12) + hw] = o[t][r] * inv_l;
        }
    }
}

extern "C" void kernel_launch(void* const* d_in, const int* in_sizes, int n_in,
                              void* d_out, int out_size, void* d_ws, size_t ws_size,
                              hipStream_t stream) {
    const float* hidden = (const float*)d_in[0];
    float* out = (float*)d_out;

    _Float16* fN = (_Float16*)d_ws;                                      // 2 MB
    _Float16* fT = (_Float16*)((char*)d_ws +
                               (size_t)N_TOK * CDIM * sizeof(_Float16)); // next 2 MB

    const int total = 2 * CDIM * 64 * 64;   // 1,048,576 elements
    tfsa_pack_kernel<<<total / 256, 256, 0, stream>>>(hidden, fN, fT);

    // 512 query tiles of 16 rows, QWAVES waves per block
    tfsa_attn_kernel<<<(N_TOK / 16) / QWAVES, 32 * QWAVES, 0, stream>>>(fN, fT, out);
}
